// CHNN_19112604467393
// MI455X (gfx1250) — matrix operand order repaired and checked by reading, not compiled, not run on hardware
//
#include <hip/hip_runtime.h>
#include <hip/hip_bf16.h>

// Problem constants (from reference): N=32 particles, D=2, BS=4096, ZD=128,
// hidden = 256, C = 32 constraints.
#define BS_   4096
#define NP    32
#define H_    256

typedef __attribute__((ext_vector_type(16))) _Float16 v16h;
typedef __attribute__((ext_vector_type(8)))  _Float16 v8h;
typedef __attribute__((ext_vector_type(8)))  float    v8f;
typedef __attribute__((ext_vector_type(4)))  int      v4i;

#define LDS_AS  __attribute__((address_space(3)))
#define GLB_AS  __attribute__((address_space(1)))

union Frag { v16h v; v8h h[2]; };

// ---------------- workspace layout (bytes) ----------------
#define OFF_R16    (0)
#define OFF_W0CM   (OFF_R16   + BS_*64*2)       // Bt of W0 (256x64)
#define OFF_W1CM   (OFF_W0CM  + 64*256*2)       // Bt of W1 (256x256)
#define OFF_W2CM   (OFF_W1CM  + 256*256*2)      // Bt of W2 (256x256)
#define OFF_W0RM   (OFF_W2CM  + 256*256*2)      // W0 row-major f16 (= Bt of W0^T)
#define OFF_W1RM   (OFF_W0RM  + 64*256*2)
#define OFF_W2RM   (OFF_W1RM  + 256*256*2)
#define OFF_H0     (OFF_W2RM  + 256*256*2)      // softplus(a0) f16
#define OFF_S0     (OFF_H0    + BS_*H_*2)       // sigmoid(a0) f16
#define OFF_H1     (OFF_S0    + BS_*H_*2)
#define OFF_S1     (OFF_H1    + BS_*H_*2)
#define OFF_S2     (OFF_S1    + BS_*H_*2)
#define OFF_GA     (OFF_S2    + BS_*H_*2)
#define OFF_GB     (OFF_GA    + BS_*H_*2)
#define OFF_DHR    (OFF_GB    + BS_*H_*2)       // dV/dr f32 (4096x64)

// -------- gfx1250 async global->LDS helpers (builtin if present, else asm) --
__device__ __forceinline__ void async_g2l_b128(const _Float16* g, LDS_AS _Float16* l) {
#if __has_builtin(__builtin_amdgcn_global_load_async_to_lds_b128)
  __builtin_amdgcn_global_load_async_to_lds_b128(
      (GLB_AS v4i*)(g), (LDS_AS v4i*)(l), 0, 0);
#else
  unsigned lo = (unsigned)(unsigned long long)l;
  unsigned long long ga = (unsigned long long)g;
  asm volatile("global_load_async_to_lds_b128 %0, %1, off"
               :: "v"(lo), "v"(ga) : "memory");
#endif
}

__device__ __forceinline__ void wait_async0() {
#if __has_builtin(__builtin_amdgcn_s_wait_asynccnt)
  __builtin_amdgcn_s_wait_asynccnt(0);
#else
  asm volatile("s_wait_asynccnt 0x0" ::: "memory");
#endif
}

// ---------------- prep kernels ----------------
__global__ void convert_w(const float* __restrict__ W, _Float16* __restrict__ Wrm,
                          _Float16* __restrict__ Wtr, int rows, int cols) {
  int i = blockIdx.x * blockDim.x + threadIdx.x;
  if (i < rows * cols) {
    int rr = i / cols, cc = i % cols;
    _Float16 v = (_Float16)W[i];
    Wrm[i] = v;                 // row-major copy (backward Bt)
    Wtr[cc * rows + rr] = v;    // transposed copy (forward Bt)
  }
}

__global__ void convert_r(const float* __restrict__ z, _Float16* __restrict__ R16) {
  int i = blockIdx.x * blockDim.x + threadIdx.x;
  if (i < BS_ * 64) {
    int s = i >> 6, j = i & 63;
    R16[i] = (_Float16)z[(size_t)s * 128 + j];
  }
}

__global__ void g2_kernel(const _Float16* __restrict__ s2, const float* __restrict__ W3,
                          _Float16* __restrict__ g2) {
  int i = blockIdx.x * blockDim.x + threadIdx.x;
  if (i < BS_ * H_) {
    int k = i & (H_ - 1);
    g2[i] = (_Float16)((float)s2[i] * W3[k]);
  }
}

// ---------------- WMMA GEMM:  C(MxN) = A(MxK) @ B(KxN)  ----------------
// A: row-major f16 (streamed from global).  BT: N-major f16 (BT[n*K+k]) —
// one block's 64-column panel is a contiguous 64*K-half slice, staged into
// LDS once via async global->LDS (ASYNCcnt), then WMMA B-fragments come from
// ds_load_b128.  LDS column stride padded +8 halves for bank spread.
// Block = 256 threads = 8 waves = 8 M-tiles x one 64-wide N-group.
// mode 0: x += bias; store softplus->outH, sigmoid->outS       (fwd hidden)
// mode 1: x += bias; store sigmoid->outS                       (fwd last)
// mode 2: store (x * S[idx]) -> outH                           (bwd masked)
// mode 3: store x -> outF (f32)                                (bwd final)
#define KPAD 8
__global__ __launch_bounds__(256)
void gemm_wmma(const _Float16* __restrict__ A, const _Float16* __restrict__ BT,
               const float* __restrict__ bias, const _Float16* __restrict__ S,
               _Float16* __restrict__ outH, _Float16* __restrict__ outS,
               float* __restrict__ outF, int M, int N, int K, int mode)
{
  __shared__ _Float16 ldsB[64 * (256 + KPAD)];

  const int lane  = threadIdx.x & 31;
  const int wave  = threadIdx.x >> 5;
  const int nG    = N >> 6;                  // 64-wide column groups
  const int ng    = blockIdx.x % nG;
  const int mtG   = blockIdx.x / nG;
  const int mBase = (mtG * 8 + wave) * 16;
  const int nBase = ng * 64;
  const int Kp    = K + KPAD;

  // ---- stage B panel (64 cols x K) into LDS via async loads ----
  LDS_AS _Float16* lds3 = (LDS_AS _Float16*)&ldsB[0];
  {
    const int kc = K >> 3;                   // 16B chunks per column
    const _Float16* gpan = BT + (size_t)nBase * K;
    for (int col = wave; col < 64; col += 8) {
      for (int w = lane; w < kc; w += 32) {
        async_g2l_b128(gpan + (size_t)col * K + w * 8,
                       lds3 + col * Kp + w * 8);
      }
    }
  }
  wait_async0();
  __syncthreads();

  const int half16 = (lane < 16) ? 0 : 1;    // lane half selects K sub-range
  const int rowA   = mBase + (lane & 15);
  const int colLo  = lane & 15;

  v8f acc[4] = {};                           // 4 accumulators: 16x64 strip

  for (int k0 = 0; k0 < K; k0 += 32) {
    // --- A fragment: 16x32 f16, per documented CDNA5 layout ---
    // lane<16 : K = k0+{0..7}, k0+{16..23} ; lane>=16 : +8 each
    Frag a;
    const _Float16* pa = A + (size_t)rowA * K + k0 + half16 * 8;
    a.h[0] = *(const v8h*)(pa);
    a.h[1] = *(const v8h*)(pa + 16);

    if (k0 + 32 < K) {
      __builtin_prefetch(pa + 32, 0, 1);
    }

    #pragma unroll
    for (int t = 0; t < 4; ++t) {
      // --- B fragment from LDS: col = 16t+colLo of panel,
      // lane<16 : K = k0+{0..7}, k0+{16..23} ; lane>=16 : +8 each
      const _Float16* pb = &ldsB[(16 * t + colLo) * Kp + k0 + half16 * 8];
      Frag b;
      b.h[0] = *(const v8h*)(pb);
      b.h[1] = *(const v8h*)(pb + 16);

      acc[t] = __builtin_amdgcn_wmma_f32_16x16x32_f16(
          /*neg_a=*/false, a.v, /*neg_b=*/false, b.v,
          /*c_mod=*/(short)0, acc[t], /*reuse_a=*/false, /*reuse_b=*/false);
    }
  }

  // --- epilogue: C layout — lane gives N=lane&15, VGPR j gives M=j+half16*8
  #pragma unroll
  for (int t = 0; t < 4; ++t) {
    const int col = nBase + 16 * t + colLo;
    float bb = 0.f;
    if (mode <= 1) bb = bias[col];
    #pragma unroll
    for (int j = 0; j < 8; ++j) {
      const int m = mBase + j + half16 * 8;
      const size_t idx = (size_t)m * N + col;
      float x = acc[t][j] + bb;
      if (mode == 0) {
        float sp = fmaxf(x, 0.f) + log1pf(__expf(-fabsf(x)));
        float sg = 1.f / (1.f + __expf(-x));
        outH[idx] = (_Float16)sp;
        outS[idx] = (_Float16)sg;
      } else if (mode == 1) {
        float sg = 1.f / (1.f + __expf(-x));
        outS[idx] = (_Float16)sg;
      } else if (mode == 2) {
        outH[idx] = (_Float16)(x * (float)S[idx]);
      } else {
        outF[idx] = x;
      }
    }
  }
}

// ---------------- analytic constrained-dynamics solve ----------------
// A = DPhi^T J DPhi = [[0, S],[−S, K]] with S = G^T M^{-1} G tridiagonal
// (chain constraints) and K = T − T^T tridiagonal antisymmetric.
// Solve reduces to two Thomas solves of size 32.  One thread per sample.
__global__ __launch_bounds__(256)
void chnn_solve(const float* __restrict__ z, const float* __restrict__ mpar,
                const float* __restrict__ dHr, float* __restrict__ out)
{
  int s = blockIdx.x * blockDim.x + threadIdx.x;
  if (s >= BS_) return;
  const float* zr = z   + (size_t)s * 128;
  const float* gr = dHr + (size_t)s * 64;

  float im[NP], r[NP][2], vv[NP][2], g[NP][2];
  #pragma unroll 1
  for (int n = 0; n < NP; ++n) {
    im[n] = __expf(-mpar[n]);
    r[n][0] = zr[2*n];      r[n][1] = zr[2*n+1];
    vv[n][0] = im[n] * zr[64+2*n];
    vv[n][1] = im[n] * zr[64+2*n+1];
    g[n][0] = gr[2*n];      g[n][1] = gr[2*n+1];
  }

  float e[NP][2], f[NP][2];
  e[0][0] = 2.f*r[0][0];  e[0][1] = 2.f*r[0][1];
  f[0][0] = 2.f*vv[0][0]; f[0][1] = 2.f*vv[0][1];
  #pragma unroll 1
  for (int c = 1; c < NP; ++c) {
    e[c][0] = 2.f*(r[c-1][0]-r[c][0]);   e[c][1] = 2.f*(r[c-1][1]-r[c][1]);
    f[c][0] = 2.f*(vv[c-1][0]-vv[c][0]); f[c][1] = 2.f*(vv[c-1][1]-vv[c][1]);
  }
  auto dot = [](const float* a, const float* b) { return a[0]*b[0] + a[1]*b[1]; };

  float dd[NP], du[NP-1], ku[NP-1], b1[NP], b2[NP];
  dd[0] = im[0] * dot(e[0], e[0]);
  du[0] = im[0] * dot(e[0], e[1]);                       // S[0,1]
  ku[0] = im[0] * (dot(f[0], e[1]) - dot(f[1], e[0]));   // K[0,1]
  #pragma unroll 1
  for (int c = 1; c < NP; ++c) {
    dd[c] = (im[c-1] + im[c]) * dot(e[c], e[c]);
    if (c < NP-1) {
      du[c] = -im[c] * dot(e[c], e[c+1]);
      ku[c] =  im[c] * (dot(f[c+1], e[c]) - dot(f[c], e[c+1]));
    }
  }
  b1[0] = 0.5f * dot(e[0], f[0]);
  b2[0] = 0.5f * dot(f[0], f[0]) - im[0] * dot(e[0], g[0]);
  #pragma unroll 1
  for (int c = 1; c < NP; ++c) {
    b1[c] = 0.5f * dot(e[c], f[c]);
    float t0 = im[c-1]*g[c-1][0] - im[c]*g[c][0];
    float t1 = im[c-1]*g[c-1][1] - im[c]*g[c][1];
    b2[c] = 0.5f * dot(f[c], f[c]) - (e[c][0]*t0 + e[c][1]*t1);
  }

  // Thomas solve #1: S x2 = b1 (S symmetric: subdiag == superdiag)
  float cp[NP-1], x2[NP], x1[NP];
  cp[0] = du[0] / dd[0];
  x2[0] = b1[0] / dd[0];
  #pragma unroll 1
  for (int c = 1; c < NP; ++c) {
    float lc = du[c-1];
    float mm = dd[c] - lc * cp[c-1];
    if (c < NP-1) cp[c] = du[c] / mm;
    x2[c] = (b1[c] - lc * x2[c-1]) / mm;
  }
  #pragma unroll 1
  for (int c = NP-2; c >= 0; --c) x2[c] -= cp[c] * x2[c+1];

  // rhs = K x2 − b2 ; Thomas solve #2: S x1 = rhs
  float rhs[NP];
  #pragma unroll 1
  for (int c = 0; c < NP; ++c) {
    float acc = -b2[c];
    if (c < NP-1) acc += ku[c]   * x2[c+1];
    if (c > 0)    acc -= ku[c-1] * x2[c-1];
    rhs[c] = acc;
  }
  x1[0] = rhs[0] / dd[0];
  #pragma unroll 1
  for (int c = 1; c < NP; ++c) {
    float lc = du[c-1];
    float mm = dd[c] - lc * cp[c-1];
    x1[c] = (rhs[c] - lc * x1[c-1]) / mm;
  }
  #pragma unroll 1
  for (int c = NP-2; c >= 0; --c) x1[c] -= cp[c] * x1[c+1];

  // out = JdH − J(DPhi X)
  float* o = out + (size_t)s * 128;
  #pragma unroll 1
  for (int n = 0; n < NP; ++n) {
    #pragma unroll
    for (int d = 0; d < 2; ++d) {
      float gx2 = (n == 0 ? x2[0]*e[0][d] : -x2[n]*e[n][d])
                + (n < NP-1 ? x2[n+1]*e[n+1][d] : 0.f);
      float gx1 = (n == 0 ? x1[0]*e[0][d] : -x1[n]*e[n][d])
                + (n < NP-1 ? x1[n+1]*e[n+1][d] : 0.f);
      float fx2 = (n == 0 ? x2[0]*f[0][d] : -x2[n]*f[n][d])
                + (n < NP-1 ? x2[n+1]*f[n+1][d] : 0.f);
      o[2*n+d]      = vv[n][d] - im[n]*gx2;        // top:    v − M⁻¹G x2
      o[64+2*n+d]   = -g[n][d] + gx1 + fx2;        // bottom: −dV/dr + Gx1 + Gd x2
    }
  }
}

// ---------------- host launch ----------------
extern "C" void kernel_launch(void* const* d_in, const int* in_sizes, int n_in,
                              void* d_out, int out_size, void* d_ws, size_t ws_size,
                              hipStream_t stream) {
  (void)in_sizes; (void)n_in; (void)out_size; (void)ws_size;
  const float* z    = (const float*)d_in[1];
  const float* mpar = (const float*)d_in[2];
  const float* W0   = (const float*)d_in[3];
  const float* b0   = (const float*)d_in[4];
  const float* W1   = (const float*)d_in[5];
  const float* b1   = (const float*)d_in[6];
  const float* W2   = (const float*)d_in[7];
  const float* b2   = (const float*)d_in[8];
  const float* W3   = (const float*)d_in[9];
  float* out = (float*)d_out;

  char* ws = (char*)d_ws;
  _Float16* R16  = (_Float16*)(ws + OFF_R16);
  _Float16* W0cm = (_Float16*)(ws + OFF_W0CM);
  _Float16* W1cm = (_Float16*)(ws + OFF_W1CM);
  _Float16* W2cm = (_Float16*)(ws + OFF_W2CM);
  _Float16* W0rm = (_Float16*)(ws + OFF_W0RM);
  _Float16* W1rm = (_Float16*)(ws + OFF_W1RM);
  _Float16* W2rm = (_Float16*)(ws + OFF_W2RM);
  _Float16* h0   = (_Float16*)(ws + OFF_H0);
  _Float16* s0   = (_Float16*)(ws + OFF_S0);
  _Float16* h1   = (_Float16*)(ws + OFF_H1);
  _Float16* s1   = (_Float16*)(ws + OFF_S1);
  _Float16* s2   = (_Float16*)(ws + OFF_S2);
  _Float16* gA   = (_Float16*)(ws + OFF_GA);
  _Float16* gB   = (_Float16*)(ws + OFF_GB);
  float*    dHr  = (float*)   (ws + OFF_DHR);

  // prep: f16 weight copies (row-major + transposed) and f16 inputs
  convert_w<<<(64*256 + 255)/256,  256, 0, stream>>>(W0, W0rm, W0cm, 64,  256);
  convert_w<<<(256*256 + 255)/256, 256, 0, stream>>>(W1, W1rm, W1cm, 256, 256);
  convert_w<<<(256*256 + 255)/256, 256, 0, stream>>>(W2, W2rm, W2cm, 256, 256);
  convert_r<<<(BS_*64 + 255)/256,  256, 0, stream>>>(z, R16);

  // forward MLP (WMMA): a = X@W + b, keep softplus + sigmoid
  gemm_wmma<<<128, 256, 0, stream>>>(R16, W0cm, b0, nullptr, h0, s0, nullptr,
                                     BS_, 256, 64,  0);
  gemm_wmma<<<128, 256, 0, stream>>>(h0,  W1cm, b1, nullptr, h1, s1, nullptr,
                                     BS_, 256, 256, 0);
  gemm_wmma<<<128, 256, 0, stream>>>(h1,  W2cm, b2, nullptr, nullptr, s2, nullptr,
                                     BS_, 256, 256, 1);

  // backward: g2 = sigmoid(a2) * W3 ; g1 = (g2 W2^T)*s1 ; g0 = (g1 W1^T)*s0 ;
  // dV/dr = g0 W0^T   (backward Bt == row-major weight copy)
  g2_kernel<<<(BS_*H_ + 255)/256, 256, 0, stream>>>(s2, W3, gA);
  gemm_wmma<<<128, 256, 0, stream>>>(gA, W2rm, nullptr, s1, gB, nullptr, nullptr,
                                     BS_, 256, 256, 2);
  gemm_wmma<<<128, 256, 0, stream>>>(gB, W1rm, nullptr, s0, gA, nullptr, nullptr,
                                     BS_, 256, 256, 2);
  gemm_wmma<<<32,  256, 0, stream>>>(gA, W0rm, nullptr, nullptr, nullptr, nullptr, dHr,
                                     BS_, 64,  256, 3);

  // analytic tridiagonal constrained solve + output assembly
  chnn_solve<<<BS_/256, 256, 0, stream>>>(z, mpar, dHr, out);
}
